// Set2Set_6846177870026
// MI455X (gfx1250) — matrix operand order repaired and checked by reading, not compiled, not run on hardware
//
#include <hip/hip_runtime.h>
#include <hip/hip_bf16.h>
#include <stdint.h>

#define NNODES 200000
#define NB     4096
#define NC     256
#define NSTEPS 4
#define KDIM   768      // [q(256) | r(256) | h(256)]
#define GDIM   1024     // 4*C gate outputs

typedef __attribute__((ext_vector_type(16))) _Float16 v16h;
typedef __attribute__((ext_vector_type(8)))  _Float16 v8h;
typedef __attribute__((ext_vector_type(8)))  float    v8f;

// ---------------- init: zero state, reset segment tables ----------------
__global__ void init_state(float* h, float* c, int* counts, int* s0, int* s1) {
  int idx = blockIdx.x * blockDim.x + threadIdx.x;
  if (idx < NB * NC) { h[idx] = 0.f; c[idx] = 0.f; }
  if (idx < NB) { counts[idx] = 0; s0[idx] = NNODES; s1[idx] = 0; }
}

// batch is sorted -> segment [s0,s1) per graph via integer atomics (deterministic)
__global__ void seg_hist(const int* __restrict__ batch, int* counts, int* s0, int* s1) {
  int n = blockIdx.x * blockDim.x + threadIdx.x;
  if (n >= NNODES) return;
  int b = batch[n];
  atomicAdd(&counts[b], 1);
  atomicMin(&s0[b], n);
  atomicMax(&s1[b], n + 1);
}

// Fold W_ih[1024x512] and W_hh[1024x256] into f16 Wc[1024x768]; bsum = b_ih+b_hh
__global__ void conv_weights(const float* __restrict__ W_ih, const float* __restrict__ W_hh,
                             const float* __restrict__ b_ih, const float* __restrict__ b_hh,
                             _Float16* __restrict__ Wc, float* __restrict__ bsum) {
  int idx = blockIdx.x * blockDim.x + threadIdx.x;
  if (idx < GDIM * KDIM) {
    int nrow = idx / KDIM, k = idx % KDIM;
    float v = (k < 2 * NC) ? W_ih[nrow * (2 * NC) + k] : W_hh[nrow * NC + (k - 2 * NC)];
    Wc[idx] = (_Float16)v;
  }
  if (idx < GDIM) bsum[idx] = b_ih[idx] + b_hh[idx];
}

// ---------------- attention + segment softmax + weighted mean ----------------
// One workgroup per graph (segments are contiguous since batch is sorted).
// 8 waves; wave w owns nodes s0+w, s0+w+8, ...; lane owns 8 channels.
__global__ __launch_bounds__(256)
void attend(const float* __restrict__ x, const float* __restrict__ h,
            const int* __restrict__ s0a, const int* __restrict__ s1a,
            const int* __restrict__ counts,
            float* __restrict__ e_ws, float* __restrict__ r_out) {
  __shared__ float qs[NC];
  __shared__ float mred[8];
  __shared__ float dred[8];
  __shared__ float rred[8 * NC];

  int b    = blockIdx.x;
  int tid  = threadIdx.x;
  int lane = tid & 31;
  int wv   = tid >> 5;

  qs[tid] = h[(size_t)b * NC + tid];   // q == previous h (zeros at step 0)
  __syncthreads();

  float q0[8];
  #pragma unroll
  for (int j = 0; j < 8; ++j) q0[j] = qs[lane * 8 + j];

  int s0 = s0a[b], s1 = s1a[b];

  // pass 1: e[n] = <x[n], q>, segment max
  float mx = -3.402823466e38f;
  for (int n = s0 + wv; n < s1; n += 8) {
    const float* xr = x + (size_t)n * NC + lane * 8;
    float e = 0.f;
    #pragma unroll
    for (int j = 0; j < 8; ++j) e += xr[j] * q0[j];
    #pragma unroll
    for (int off = 16; off >= 1; off >>= 1) e += __shfl_xor(e, off, 32);
    if (lane == 0) e_ws[n] = e;
    mx = fmaxf(mx, e);
  }
  if (lane == 0) mred[wv] = mx;
  __syncthreads();
  if (tid == 0) {
    float m = mred[0];
    #pragma unroll
    for (int w = 1; w < 8; ++w) m = fmaxf(m, mred[w]);
    mred[0] = m;
  }
  __syncthreads();
  float m = mred[0];

  // pass 2: denom += exp(e-m); racc += exp(e-m)*x
  float racc[8];
  #pragma unroll
  for (int j = 0; j < 8; ++j) racc[j] = 0.f;
  float dsum = 0.f;
  for (int n = s0 + wv; n < s1; n += 8) {
    float wgt = expf(e_ws[n] - m);
    dsum += wgt;
    const float* xr = x + (size_t)n * NC + lane * 8;
    #pragma unroll
    for (int j = 0; j < 8; ++j) racc[j] += wgt * xr[j];
  }
  #pragma unroll
  for (int j = 0; j < 8; ++j) rred[wv * NC + lane * 8 + j] = racc[j];
  if (lane == 0) dred[wv] = dsum;
  __syncthreads();

  float denom = 0.f, rsum = 0.f;
  #pragma unroll
  for (int w = 0; w < 8; ++w) { denom += dred[w]; rsum += rred[w * NC + tid]; }
  float cnt = fmaxf((float)counts[b], 1.f);
  r_out[(size_t)b * NC + tid] = rsum / (fmaxf(denom, 1e-16f) * cnt);
}

// ---------------- pack u = [h, r, h] as f16 ----------------
__global__ void pack_u(const float* __restrict__ h, const float* __restrict__ r,
                       _Float16* __restrict__ u) {
  int idx = blockIdx.x * blockDim.x + threadIdx.x;
  if (idx >= NB * KDIM) return;
  int b = idx / KDIM, k = idx % KDIM;
  float v = (k < NC) ? h[b * NC + k]
          : (k < 2 * NC) ? r[b * NC + (k - NC)]
          : h[b * NC + (k - 2 * NC)];
  u[idx] = (_Float16)v;
}

// ---------------- WMMA GEMM: gates[4096x1024] = u[4096x768] * Wc^T + bsum ----------------
// One wave per 16x16 output tile, 24 x v_wmma_f32_16x16x32_f16 along K.
__global__ __launch_bounds__(256)
void gemm_gates(const _Float16* __restrict__ u, const _Float16* __restrict__ Wc,
                const float* __restrict__ bsum, float* __restrict__ gates) {
  int lane = threadIdx.x & 31;
  int wv   = threadIdx.x >> 5;
  int m0   = blockIdx.x * 16;                 // batch-row tile
  int n0   = (blockIdx.y * 8 + wv) * 16;      // gate-column tile
  int row  = lane & 15;
  int hi   = lane >> 4;

  // C/D layout: acc[r] = elem[M=m0+r+8*hi][N=n0+row]; bias depends on N only.
  float bv = bsum[n0 + row];
  v8f acc;
  #pragma unroll
  for (int j = 0; j < 8; ++j) acc[j] = bv;

  // A (16-bit 16x32): lane<16 -> row M=lane, K 0..15; lane>=16 -> K 16..31
  const _Float16* aBase = u  + (size_t)(m0 + row) * KDIM + hi * 16;
  // B (32x16): lane<16 -> col N=lane, K 0..15; lane>=16 -> K 16..31. B[k][n] = Wc[n][k].
  const _Float16* bBase = Wc + (size_t)(n0 + row) * KDIM + hi * 16;

  #pragma unroll
  for (int k0 = 0; k0 < KDIM; k0 += 32) {
    v16h a = *(const v16h*)(aBase + k0);       // K = k0 + hi*16 + [0..15]
    v16h bm = *(const v16h*)(bBase + k0);      // 32B aligned (KDIM*2 and offsets % 32 == 0)
    acc = __builtin_amdgcn_wmma_f32_16x16x32_f16(
        /*neg_a=*/false, a, /*neg_b=*/false, bm,
        /*c_mod=*/(short)0, acc, /*reuse_a=*/false, /*reuse_b=*/false);
  }

  float* gp = gates + (size_t)(m0 + hi * 8) * GDIM + n0 + row;
  #pragma unroll
  for (int r = 0; r < 8; ++r) gp[(size_t)r * GDIM] = acc[r];
}

// ---------------- LSTM pointwise: c,h update (torch gate order i,f,g,o) ----------------
__global__ void lstm_pointwise(const float* __restrict__ gates,
                               float* __restrict__ cst, float* __restrict__ h) {
  int idx = blockIdx.x * blockDim.x + threadIdx.x;
  if (idx >= NB * NC) return;
  int b = idx / NC, ch = idx % NC;
  const float* g = gates + (size_t)b * GDIM;
  float ig = g[ch], fg = g[NC + ch], gg = g[2 * NC + ch], og = g[3 * NC + ch];
  float si = 1.f / (1.f + expf(-ig));
  float sf = 1.f / (1.f + expf(-fg));
  float so = 1.f / (1.f + expf(-og));
  float cn = sf * cst[idx] + si * tanhf(gg);
  cst[idx] = cn;
  h[idx]   = so * tanhf(cn);
}

// ---------------- output: [q, r] ----------------
__global__ void finalize(const float* __restrict__ h, const float* __restrict__ r,
                         float* __restrict__ out) {
  int idx = blockIdx.x * blockDim.x + threadIdx.x;
  if (idx >= NB * 2 * NC) return;
  int b = idx / (2 * NC), c2 = idx % (2 * NC);
  out[idx] = (c2 < NC) ? h[(size_t)b * NC + c2] : r[(size_t)b * NC + (c2 - NC)];
}

extern "C" void kernel_launch(void* const* d_in, const int* in_sizes, int n_in,
                              void* d_out, int out_size, void* d_ws, size_t ws_size,
                              hipStream_t stream) {
  const float* x     = (const float*)d_in[0];
  const int*   batch = (const int*)d_in[1];
  const float* W_ih  = (const float*)d_in[2];
  const float* W_hh  = (const float*)d_in[3];
  const float* b_ih  = (const float*)d_in[4];
  const float* b_hh  = (const float*)d_in[5];
  float* out = (float*)d_out;

  // workspace carve-up (256B aligned)
  char* ws = (char*)d_ws;
  size_t off = 0;
  auto carve = [&](size_t bytes) -> void* {
    void* p = ws + off;
    off = (off + bytes + 255) & ~(size_t)255;
    return p;
  };
  float*     h_buf  = (float*)     carve((size_t)NB * NC * 4);
  float*     c_buf  = (float*)     carve((size_t)NB * NC * 4);
  float*     r_buf  = (float*)     carve((size_t)NB * NC * 4);
  float*     gates  = (float*)     carve((size_t)NB * GDIM * 4);
  _Float16*  u16    = (_Float16*)  carve((size_t)NB * KDIM * 2);
  _Float16*  Wc     = (_Float16*)  carve((size_t)GDIM * KDIM * 2);
  float*     bsum   = (float*)     carve((size_t)GDIM * 4);
  float*     e_ws   = (float*)     carve((size_t)NNODES * 4);
  int*       counts = (int*)       carve((size_t)NB * 4);
  int*       seg0   = (int*)       carve((size_t)NB * 4);
  int*       seg1   = (int*)       carve((size_t)NB * 4);
  (void)ws_size; (void)in_sizes; (void)n_in; (void)out_size;

  init_state<<<(NB * NC + 255) / 256, 256, 0, stream>>>(h_buf, c_buf, counts, seg0, seg1);
  seg_hist<<<(NNODES + 255) / 256, 256, 0, stream>>>(batch, counts, seg0, seg1);
  conv_weights<<<(GDIM * KDIM + 255) / 256, 256, 0, stream>>>(W_ih, W_hh, b_ih, b_hh, Wc, bsum);

  for (int step = 0; step < NSTEPS; ++step) {
    attend<<<NB, 256, 0, stream>>>(x, h_buf, seg0, seg1, counts, e_ws, r_buf);
    pack_u<<<(NB * KDIM + 255) / 256, 256, 0, stream>>>(h_buf, r_buf, u16);
    gemm_gates<<<dim3(NB / 16, GDIM / (16 * 8)), 256, 0, stream>>>(u16, Wc, bsum, gates);
    lstm_pointwise<<<(NB * NC + 255) / 256, 256, 0, stream>>>(gates, c_buf, h_buf);
  }

  finalize<<<(NB * 2 * NC + 255) / 256, 256, 0, stream>>>(h_buf, r_buf, out);
}
